// BasicBlock_21131239097114
// MI455X (gfx1250) — matrix operand order repaired and checked by reading, not compiled, not run on hardware
//
#include <hip/hip_runtime.h>

// ---------------- problem constants ----------------
#define NB    32
#define CH    128
#define HH    28
#define HW    784           // 28*28
#define S_TOT 25088         // NB*HW
#define K9    1152          // CH*3*3
#define KS    1160          // padded im2col K-stride (2320B, 16B aligned)
#define TOT   3211264       // NB*CH*HW
#define NCHUNK 36           // K9 / 32
#define BSTRIDE 40          // LDS B-tile row stride in bf16 (80B: 16B aligned, bank-spread)

typedef __bf16 v8bf  __attribute__((ext_vector_type(8)));
typedef __bf16 v16bf __attribute__((ext_vector_type(16)));
typedef float  v8f   __attribute__((ext_vector_type(8)));
typedef int    v4i_vs __attribute__((vector_size(16)));   // builtin's expected pointee type

#if defined(__has_builtin)
#  if __has_builtin(__builtin_amdgcn_global_load_async_to_lds_b128) && \
      __has_builtin(__builtin_amdgcn_s_wait_asynccnt)
#    define USE_ASYNC_LDS 1
#  else
#    define USE_ASYNC_LDS 0
#  endif
#else
#  define USE_ASYNC_LDS 0
#endif

// ---------------- 1) power-of-two quantize -> bf16 (exact for 2^k) ----------------
__global__ void quant_bf16_kernel(const float* __restrict__ w, __bf16* __restrict__ out, int n) {
    int i = blockIdx.x * blockDim.x + threadIdx.x;
    if (i >= n) return;
    float v = w[i];
    float aw = fabsf(v);
    float q = 0.0f;
    if (aw >= 0.005f)
        q = copysignf(exp2f(rintf(log2f(fmaxf(aw, 1e-10f)))), v);   // rintf == round-half-even
    out[i] = (__bf16)q;
}

// ---------------- 2) im2col NCHW fp32 -> P[s][k] bf16 (K contiguous per position) ----------------
__global__ __launch_bounds__(256) void im2col_kernel(const float* __restrict__ src,
                                                     __bf16* __restrict__ P) {
    int s = blockIdx.x;                       // 0..S_TOT-1 = (n,h,w)
    int n = s / HW;  int hw = s - n * HW;
    int h = hw / HH; int w = hw - h * HH;
    const float* base = src + (size_t)n * CH * HW;
    __bf16* dst = P + (size_t)s * KS;
    for (int k = threadIdx.x; k < K9; k += blockDim.x) {
        int ci = k / 9; int t = k - ci * 9;
        int r = t / 3;  int c = t - r * 3;
        int ih = h + r - 1, iw = w + c - 1;
        float v = 0.0f;
        if ((unsigned)ih < HH && (unsigned)iw < HH)
            v = base[ci * HW + ih * HH + iw];
        dst[k] = (__bf16)v;
    }
}

// ---------------- 3) shift conv as WMMA GEMM with async-LDS double-buffered B ----------------
// block = 8 waves; wave w owns Cout tile w*16..w*16+15; block owns 64 spatial cols.
// B tile (64 cols x 32 k, 4KB) staged once per block per chunk (vs 8x duplicated
// global reads); async copy of chunk i+1 overlaps WMMAs of chunk i.
__global__ __launch_bounds__(256) void shift_gemm_kernel(const __bf16* __restrict__ P,
                                                         const __bf16* __restrict__ Wq,
                                                         float* __restrict__ y) {
    __shared__ __bf16 btile[2][64 * BSTRIDE];        // 10,240 B
    int lane = threadIdx.x & 31;
    int wave = threadIdx.x >> 5;              // co tile 0..7
    int s0   = blockIdx.x * 64;
    int m    = lane & 15;                     // A row / B col within 16-tile
    int hi   = lane >> 4;
    const __bf16* wrow = Wq + (size_t)(wave * 16 + m) * K9;

    // copy role: thread t moves one 16B slice of the 4KB B tile per chunk
    int ccol  = threadIdx.x >> 2;             // 0..63 : spatial column
    int cpart = threadIdx.x & 3;              // 0..3  : 8-element k group
    const __bf16* gsrc = P + (size_t)(s0 + ccol) * KS + cpart * 8;
    __bf16* lds_slot0 = &btile[0][ccol * BSTRIDE + cpart * 8];
    __bf16* lds_slot1 = &btile[1][ccol * BSTRIDE + cpart * 8];

    v8f acc[4] = {};

    auto issue_copy = [&](int chunk) {
        const __bf16* g = gsrc + chunk * 32;
        __bf16* l = (chunk & 1) ? lds_slot1 : lds_slot0;
#if USE_ASYNC_LDS
        __builtin_amdgcn_global_load_async_to_lds_b128(
            (__attribute__((address_space(1))) v4i_vs*)(g),
            (__attribute__((address_space(3))) v4i_vs*)(l),
            0, 0);
#else
        *(v8bf*)l = *(const v8bf*)g;
#endif
    };

    issue_copy(0);
    for (int ch = 0; ch < NCHUNK; ++ch) {
        int kc = ch * 32;
        if (ch + 1 < NCHUNK) issue_copy(ch + 1);
#if USE_ASYNC_LDS
        if (ch + 1 < NCHUNK) __builtin_amdgcn_s_wait_asynccnt(1);  // chunk ch landed (in-order)
        else                 __builtin_amdgcn_s_wait_asynccnt(0);
#endif
        __syncthreads();                       // chunk ch fully in LDS, all waves

        // A fragment (16x32): per lane two contiguous 8-element K runs from global
        v8bf alo = *(const v8bf*)(wrow + kc + hi * 8);
        v8bf ahi = *(const v8bf*)(wrow + kc + 16 + hi * 8);
        v16bf afrag;
        #pragma unroll
        for (int e = 0; e < 8; ++e) { afrag[e] = alo[e]; afrag[e + 8] = ahi[e]; }
        if (kc + 32 < K9)
            __builtin_prefetch(wrow + kc + 32, 0, 1);   // global_prefetch_b8

        const __bf16* buf = (ch & 1) ? &btile[1][0] : &btile[0][0];
        #pragma unroll
        for (int ct = 0; ct < 4; ++ct) {
            const __bf16* lp = buf + (ct * 16 + m) * BSTRIDE + hi * 8;
            v8bf blo = *(const v8bf*)(lp);
            v8bf bhi = *(const v8bf*)(lp + 16);
            v16bf bfrag;
            #pragma unroll
            for (int e = 0; e < 8; ++e) { bfrag[e] = blo[e]; bfrag[e + 8] = bhi[e]; }
            acc[ct] = __builtin_amdgcn_wmma_f32_16x16x32_bf16(
                false, afrag, false, bfrag, (short)0, acc[ct], false, false);
        }
        __syncthreads();                       // all waves done with buf before overwrite
    }

    // D layout: VGPR r -> M = r + 8*hi, N = m
    #pragma unroll
    for (int ct = 0; ct < 4; ++ct) {
        int s = s0 + ct * 16 + m;
        int n = s / HW, hw = s - n * HW;
        #pragma unroll
        for (int r = 0; r < 8; ++r) {
            int co = wave * 16 + r + hi * 8;
            y[((size_t)n * CH + co) * HW + hw] = acc[ct][r];
        }
    }
}

// ---------------- 4) AdderNet conv: out[co,s] = -sum_k |patch - w| ----------------
// one block per (n,h); 3 halo rows x 128 Cin staged in LDS; thread = 2 Cout x 7 w.
__global__ __launch_bounds__(256) void adder_conv_kernel(const float* __restrict__ y,
                                                         const float* __restrict__ Wa,
                                                         float* __restrict__ out) {
    __shared__ float tile[3][CH][32];         // slot j <-> input w = j-1 (halo), 48KB
    int b = blockIdx.x;
    int n = b / HH, h = b - n * HH;
    const float* ybase = y + (size_t)n * CH * HW;
    for (int idx = threadIdx.x; idx < 3 * CH * 32; idx += blockDim.x) {
        int r = idx / (CH * 32); int rem = idx - r * (CH * 32);
        int ci = rem >> 5; int j = rem & 31;
        int ih = h + r - 1, iw = j - 1;
        float v = 0.0f;
        if ((unsigned)ih < HH && (unsigned)iw < HH)
            v = ybase[ci * HW + ih * HH + iw];
        tile[r][ci][j] = v;
    }
    __syncthreads();

    int cop = threadIdx.x >> 2;               // 0..63 -> Cout pair
    int wq  = threadIdx.x & 3;                // 0..3  -> group of 7 w
    int co0 = cop * 2;
    int w0  = wq * 7;
    float acc0[7] = {}, acc1[7] = {};
    const float* wp0 = Wa + (size_t)co0 * K9;
    const float* wp1 = wp0 + K9;

    for (int ci = 0; ci < CH; ++ci) {
        #pragma unroll
        for (int r = 0; r < 3; ++r) {
            float p[9];
            #pragma unroll
            for (int j = 0; j < 9; ++j) p[j] = tile[r][ci][w0 + j];
            int kb = ci * 9 + r * 3;
            float wa0 = wp0[kb], wa1 = wp0[kb + 1], wa2 = wp0[kb + 2];
            float wb0 = wp1[kb], wb1 = wp1[kb + 1], wb2 = wp1[kb + 2];
            #pragma unroll
            for (int i = 0; i < 7; ++i) {
                acc0[i] += fabsf(p[i] - wa0) + fabsf(p[i + 1] - wa1) + fabsf(p[i + 2] - wa2);
                acc1[i] += fabsf(p[i] - wb0) + fabsf(p[i + 1] - wb1) + fabsf(p[i + 2] - wb2);
            }
        }
    }
    float* o0 = out + ((size_t)n * CH + co0) * HW + h * HH + w0;
    float* o1 = o0 + HW;
    #pragma unroll
    for (int i = 0; i < 7; ++i) { o0[i] = -acc0[i]; o1[i] = -acc1[i]; }
}

// ---------------- 5) BN batch stats per channel ----------------
__global__ __launch_bounds__(256) void bn_stats_kernel(const float* __restrict__ a,
                                                       float* __restrict__ mean,
                                                       float* __restrict__ istd) {
    __shared__ float s_sum[256], s_sq[256];
    int c = blockIdx.x;
    float sum = 0.0f, sq = 0.0f;
    for (int i = threadIdx.x; i < NB * HW; i += 256) {
        int n = i / HW, hw = i - n * HW;
        float v = a[((size_t)n * CH + c) * HW + hw];
        sum += v; sq += v * v;
    }
    s_sum[threadIdx.x] = sum; s_sq[threadIdx.x] = sq;
    __syncthreads();
    for (int off = 128; off > 0; off >>= 1) {
        if (threadIdx.x < off) {
            s_sum[threadIdx.x] += s_sum[threadIdx.x + off];
            s_sq[threadIdx.x]  += s_sq[threadIdx.x + off];
        }
        __syncthreads();
    }
    if (threadIdx.x == 0) {
        const float inv = 1.0f / (float)(NB * HW);
        float m   = s_sum[0] * inv;
        float var = s_sq[0] * inv - m * m;
        mean[c] = m;
        istd[c] = rsqrtf(var + 1e-5f);
    }
}

// ---------------- 6) fused BN apply + optional residual + ReLU ----------------
__global__ void bn_apply_kernel(const float* __restrict__ a, const float* __restrict__ gamma,
                                const float* __restrict__ beta, const float* __restrict__ mean,
                                const float* __restrict__ istd, const float* __restrict__ residual,
                                float* __restrict__ out) {
    int i = blockIdx.x * blockDim.x + threadIdx.x;
    if (i >= TOT) return;
    int c = (i / HW) % CH;
    float v = (a[i] - mean[c]) * istd[c] * gamma[c] + beta[c];
    if (residual) v += residual[i];
    out[i] = fmaxf(v, 0.0f);
}

// ---------------- host orchestration ----------------
extern "C" void kernel_launch(void* const* d_in, const int* in_sizes, int n_in,
                              void* d_out, int out_size, void* d_ws, size_t ws_size,
                              hipStream_t stream) {
    const float* x        = (const float*)d_in[0];
    const float* w_shift1 = (const float*)d_in[1];
    const float* w_add1   = (const float*)d_in[2];
    const float* g1       = (const float*)d_in[3];
    const float* b1       = (const float*)d_in[4];
    const float* w_shift2 = (const float*)d_in[5];
    const float* w_add2   = (const float*)d_in[6];
    const float* g2       = (const float*)d_in[7];
    const float* b2       = (const float*)d_in[8];
    float* out = (float*)d_out;

    char* ws = (char*)d_ws;
    size_t off = 0;
    __bf16* P   = (__bf16*)(ws + off); off += (size_t)S_TOT * KS * 2;   // 58.2 MB (L2-resident)
    float*  y   = (float*)(ws + off);  off += (size_t)TOT * 4;
    float*  a   = (float*)(ws + off);  off += (size_t)TOT * 4;
    float*  act = (float*)(ws + off);  off += (size_t)TOT * 4;
    __bf16* wq1 = (__bf16*)(ws + off); off += (size_t)CH * K9 * 2;
    __bf16* wq2 = (__bf16*)(ws + off); off += (size_t)CH * K9 * 2;
    float*  st  = (float*)(ws + off);  // mean1[128], istd1[128], mean2[128], istd2[128]

    const int NW = CH * K9;
    quant_bf16_kernel<<<(NW + 255) / 256, 256, 0, stream>>>(w_shift1, wq1, NW);
    quant_bf16_kernel<<<(NW + 255) / 256, 256, 0, stream>>>(w_shift2, wq2, NW);

    // ---- block 1 ----
    im2col_kernel<<<S_TOT, 256, 0, stream>>>(x, P);
    shift_gemm_kernel<<<S_TOT / 64, 256, 0, stream>>>(P, wq1, y);
    adder_conv_kernel<<<NB * HH, 256, 0, stream>>>(y, w_add1, a);
    bn_stats_kernel<<<CH, 256, 0, stream>>>(a, st, st + 128);
    bn_apply_kernel<<<(TOT + 255) / 256, 256, 0, stream>>>(a, g1, b1, st, st + 128, nullptr, act);

    // ---- block 2 ----
    im2col_kernel<<<S_TOT, 256, 0, stream>>>(act, P);
    shift_gemm_kernel<<<S_TOT / 64, 256, 0, stream>>>(P, wq2, y);
    adder_conv_kernel<<<NB * HH, 256, 0, stream>>>(y, w_add2, a);
    bn_stats_kernel<<<CH, 256, 0, stream>>>(a, st + 256, st + 384);
    bn_apply_kernel<<<(TOT + 255) / 256, 256, 0, stream>>>(a, g2, b2, st + 256, st + 384, x, out);
}
